// GATTimeEncoder_2284922601632
// MI455X (gfx1250) — matrix operand order repaired and checked by reading, not compiled, not run on hardware
//
#include <hip/hip_runtime.h>

typedef _Float16 f16;
typedef __attribute__((ext_vector_type(16))) _Float16 v16h;
typedef __attribute__((ext_vector_type(8)))  float    v8f;

#define N_NODES 50000
#define N_PAD   50048                 // 782 * 64 : GEMM row padding
#define E_EDGES 800000
#define EF      (E_EDGES + N_NODES)   // 850000 (real edges + self loops)
#define HID     128
#define NEG_SLOPE 0.2f

// ---------------------------------------------------------------------------
// Fragment-order permutations (so WMMA operands load as contiguous b128s).
// A 16x32 f16 wmma layout: lane<16 row M=lane holds K {0..7,16..23}; lane>=16
// same row, K {8..15,24..31}. We store each X row's 128 K-values permuted so a
// lane's 16 halves for k-step t sit at row*128 + t*32 + hi*16 + [0..15].
// ---------------------------------------------------------------------------
__device__ __forceinline__ int a_frag_off(int k) {
  int t  = k >> 5;
  int rem = k & 31;
  int hi = (rem >> 3) & 1;                 // which half-wave owns this K
  int h  = (rem & 7) | ((rem & 16) >> 1);  // +8 for the second K-group
  return t * 32 + hi * 16 + h;
}
// B 32x16: lane n holds K {0..7,16..23} (hi=0) or {8..15,24..31} (hi=1) for
// k-step t, same order as A. Stored contiguous: ((t*128 + col)*2 + hi)*16 + h.
__device__ __forceinline__ int b_frag_off(int k, int col) {
  int t  = k >> 5;
  int g  = (k >> 3) & 1;
  int h  = (k & 7) | ((k & 16) >> 1);
  return ((t * 128 + col) * 2 + g) * 16 + h;
}

// ---------------------------------------------------------------------------
// Per-layer init: acc = bias broadcast, lmax=-inf, denom=0, deg/easum=0,
// optional f32->f16 conversion of X into A-fragment order.
// ---------------------------------------------------------------------------
__global__ void gat_init(const float* __restrict__ xin, f16* __restrict__ xh,
                         const float* __restrict__ bias, float* __restrict__ acc,
                         float* __restrict__ lmax, float* __restrict__ denom,
                         float* __restrict__ deg, float* __restrict__ easum,
                         int convert_x) {
  int tid = blockIdx.x * blockDim.x + threadIdx.x;
  if (tid >= N_NODES * HID) return;
  acc[tid] = bias[tid & (HID - 1)];
  if (convert_x) {
    int r = tid >> 7, k = tid & 127;
    xh[r * HID + a_frag_off(k)] = (f16)xin[tid];
  }
  if (tid < N_NODES) { deg[tid] = 0.f; easum[tid] = 0.f; }
  if (tid < N_NODES * 4) { lmax[tid] = -3.0e38f; denom[tid] = 0.f; }
}

// deg[d] += 1, easum[d] += ew  (self-loop edge attr = easum/max(deg,1))
__global__ void gat_degree(const int* __restrict__ dst, const float* __restrict__ ew,
                           float* __restrict__ deg, float* __restrict__ easum) {
  int e = blockIdx.x * blockDim.x + threadIdx.x;
  if (e >= E_EDGES) return;
  int d = dst[e];
  atomicAdd(&deg[d], 1.0f);
  atomicAdd(&easum[d], ew[e]);
}

// f32 weights -> f16 copies in B-fragment order (one-shot per layer)
__global__ void gat_convw(const float* __restrict__ Wl, const float* __restrict__ Wr,
                          f16* __restrict__ WlH, f16* __restrict__ WrH) {
  int t = blockIdx.x * blockDim.x + threadIdx.x;
  if (t >= 2 * HID * HID) return;
  int idx = (t < HID * HID) ? t : t - HID * HID;
  int k = idx >> 7, col = idx & 127;
  if (t < HID * HID) WlH[b_frag_off(k, col)] = (f16)Wl[idx];
  else               WrH[b_frag_off(k, col)] = (f16)Wr[idx];
}

// ---------------------------------------------------------------------------
// WMMA GEMM:  out = X(64-row block) @ W(128x128) + bias
// blockIdx.x = 64-row block, blockIdx.y selects (Wl->xl) vs (Wr->xr).
// 8 waves; wave w owns columns [16w,16w+16) over 4 row tiles.
// xh/xl/xr are padded to N_PAD = 782*64 rows, so there are NO bounds checks:
// straight-line fills, 16 WMMAs, and unguarded stores at immediate offsets.
// ---------------------------------------------------------------------------
__global__ __launch_bounds__(256) void gat_gemm_wmma(
    const f16* __restrict__ Xf,
    const f16* __restrict__ WfA, const f16* __restrict__ WfB,
    const float* __restrict__ bA, const float* __restrict__ bB,
    float* __restrict__ outA, float* __restrict__ outB) {
  __shared__ __align__(16) f16 Wsf[HID * HID];  // 32 KB, fragment order
  __shared__ __align__(16) f16 Xsf[64 * HID];   // 16 KB, fragment order

  const int tid = threadIdx.x;
  const int rowbase = blockIdx.x * 64;
  const bool second = (blockIdx.y != 0);
  const f16*   Wf   = second ? WfB : WfA;
  const float* bias = second ? bB : bA;
  float*       out  = second ? outB : outA;

  // Cooperative fills (layout-preserving, fully vectorized, no guards)
  {
    const float4* Wv = (const float4*)Wf;
    float4* Wsv = (float4*)Wsf;
#pragma unroll
    for (int i = 0; i < 8; ++i) Wsv[tid + 256 * i] = Wv[tid + 256 * i];

    const float4* Xv = (const float4*)(Xf + (size_t)rowbase * HID);
    float4* Xsv = (float4*)Xsf;
#pragma unroll
    for (int i = 0; i < 4; ++i) Xsv[tid + 256 * i] = Xv[tid + 256 * i];
  }
  __syncthreads();

  const int lane = tid & 31;
  const int wave = tid >> 5;
  const int col0 = wave * 16;
  const int hi   = (lane >> 4) & 1;
  const int m    = lane & 15;
  const int n    = lane & 15;

  // Hoist 4 B fragments (K=128) for this column tile
  v16h bfr[4];
#pragma unroll
  for (int t = 0; t < 4; ++t) {
    const f16* bp = Wsf + ((t * 128 + col0 + n) * 2 + hi) * 16;
    ((float4*)&bfr[t])[0] = *(const float4*)(bp);
    ((float4*)&bfr[t])[1] = *(const float4*)(bp + 8);
  }
  const float bv = bias[col0 + n];

#pragma unroll
  for (int rt = 0; rt < 4; ++rt) {
    v8f c = {};
#pragma unroll
    for (int t = 0; t < 4; ++t) {
      const f16* ap = Xsf + rt * (16 * HID) + m * HID + t * 32 + hi * 16;
      v16h a;
      ((float4*)&a)[0] = *(const float4*)(ap);
      ((float4*)&a)[1] = *(const float4*)(ap + 8);
      c = __builtin_amdgcn_wmma_f32_16x16x32_f16(
          /*neg_a=*/false, a, /*neg_b=*/false, bfr[t],
          /*c_mod=*/(short)0, c, /*reuse_a=*/false, /*reuse_b=*/false);
    }
    const int row0 = rowbase + rt * 16 + (hi ? 8 : 0);  // lane's first C row
    float* op = out + (size_t)row0 * HID + col0 + n;
#pragma unroll
    for (int r = 0; r < 8; ++r) op[r * HID] = c[r] + bv;
  }
}

// ---------------------------------------------------------------------------
// Pass 1: one wave per (real or self-loop) edge. m = xl[s]+xr[d]+ea*We,
// leaky-relu, per-head dot with att -> logits; atomic segment max over dst.
// ---------------------------------------------------------------------------
__global__ __launch_bounds__(256) void gat_logits_max(
    const int* __restrict__ src, const int* __restrict__ dst,
    const float* __restrict__ ew, const float* __restrict__ deg,
    const float* __restrict__ easum,
    const float* __restrict__ xl, const float* __restrict__ xr,
    const float* __restrict__ We, const float* __restrict__ att,
    float* __restrict__ logits, float* __restrict__ lmax) {
  int e = blockIdx.x * 8 + (threadIdx.x >> 5);
  if (e >= EF) return;
  int lane = threadIdx.x & 31;
  int s, d; float ea;
  if (e < E_EDGES) { s = src[e]; d = dst[e]; ea = ew[e]; }
  else { int i = e - E_EDGES; s = i; d = i; ea = easum[i] / fmaxf(deg[i], 1.0f); }

  int c0 = lane * 4;                       // 4 channels/lane, head = lane>>3
  float4 a4 = *(const float4*)(xl + (size_t)s * HID + c0);
  float4 b4 = *(const float4*)(xr + (size_t)d * HID + c0);
  float4 w4 = *(const float4*)(We + c0);
  float4 t4 = *(const float4*)(att + c0);  // att (4,32) flat == channel index

  float part = 0.f, mv;
  mv = a4.x + b4.x + ea * w4.x; mv = mv >= 0.f ? mv : NEG_SLOPE * mv; part += mv * t4.x;
  mv = a4.y + b4.y + ea * w4.y; mv = mv >= 0.f ? mv : NEG_SLOPE * mv; part += mv * t4.y;
  mv = a4.z + b4.z + ea * w4.z; mv = mv >= 0.f ? mv : NEG_SLOPE * mv; part += mv * t4.z;
  mv = a4.w + b4.w + ea * w4.w; mv = mv >= 0.f ? mv : NEG_SLOPE * mv; part += mv * t4.w;

  part += __shfl_xor(part, 1, 32);         // reduce 8 lanes of a head
  part += __shfl_xor(part, 2, 32);
  part += __shfl_xor(part, 4, 32);

  if ((lane & 7) == 0) {
    int h = lane >> 3;
    logits[(size_t)e * 4 + h] = part;
    atomicMax(&lmax[d * 4 + h], part);
  }
}

// Pass 2: p = exp(logit - lmax[dst]); denom[dst] += p (in-place over logits)
__global__ void gat_expsum(const int* __restrict__ dst,
                           const float* __restrict__ lmax,
                           float* __restrict__ logits, float* __restrict__ denom) {
  long tid = (long)blockIdx.x * blockDim.x + threadIdx.x;
  if (tid >= (long)EF * 4) return;
  int e = (int)(tid >> 2), h = (int)(tid & 3);
  int d = (e < E_EDGES) ? dst[e] : (e - E_EDGES);
  float p = __expf(logits[tid] - lmax[d * 4 + h]);
  logits[tid] = p;
  atomicAdd(&denom[d * 4 + h], p);
}

// Pass 3: acc[dst] += (p/denom) * xl[src]   (one wave per edge, 4 ch/lane)
__global__ __launch_bounds__(256) void gat_aggregate(
    const int* __restrict__ src, const int* __restrict__ dst,
    const float* __restrict__ p, const float* __restrict__ denom,
    const float* __restrict__ xl, float* __restrict__ acc) {
  int e = blockIdx.x * 8 + (threadIdx.x >> 5);
  if (e >= EF) return;
  int lane = threadIdx.x & 31;
  int s, d;
  if (e < E_EDGES) { s = src[e]; d = dst[e]; }
  else { s = d = e - E_EDGES; }
  int c0 = lane * 4;
  int h = lane >> 3;
  float coef = p[(size_t)e * 4 + h] / denom[d * 4 + h];
  float4 a4 = *(const float4*)(xl + (size_t)s * HID + c0);
  float* ap = acc + (size_t)d * HID + c0;
  atomicAdd(ap + 0, coef * a4.x);
  atomicAdd(ap + 1, coef * a4.y);
  atomicAdd(ap + 2, coef * a4.z);
  atomicAdd(ap + 3, coef * a4.w);
}

// PReLU between layers + f16 convert into A-fragment order for next GEMM
__global__ void gat_prelu_conv(const float* __restrict__ acc,
                               const float* __restrict__ pa, f16* __restrict__ xh) {
  int tid = blockIdx.x * blockDim.x + threadIdx.x;
  if (tid >= N_NODES * HID) return;
  float a = pa[0];
  float v = acc[tid];
  v = v >= 0.f ? v : a * v;
  int r = tid >> 7, k = tid & 127;
  xh[r * HID + a_frag_off(k)] = (f16)v;
}

// ---------------------------------------------------------------------------
extern "C" void kernel_launch(void* const* d_in, const int* in_sizes, int n_in,
                              void* d_out, int out_size, void* d_ws, size_t ws_size,
                              hipStream_t stream) {
  (void)in_sizes; (void)n_in; (void)out_size; (void)ws_size;

  const float* x   = (const float*)d_in[0];
  const int*   ei  = (const int*)d_in[1];
  const int*   src = ei;
  const int*   dst = ei + E_EDGES;
  const float* ew  = (const float*)d_in[2];
  const float* Wl[2]  = {(const float*)d_in[3],  (const float*)d_in[10]};
  const float* bl[2]  = {(const float*)d_in[4],  (const float*)d_in[11]};
  const float* Wr[2]  = {(const float*)d_in[5],  (const float*)d_in[12]};
  const float* br[2]  = {(const float*)d_in[6],  (const float*)d_in[13]};
  const float* We[2]  = {(const float*)d_in[7],  (const float*)d_in[14]};
  const float* att[2] = {(const float*)d_in[8],  (const float*)d_in[15]};
  const float* bb[2]  = {(const float*)d_in[9],  (const float*)d_in[16]};
  const float* prelu_a = (const float*)d_in[17];

  // Carve workspace (xh/xl/xr padded to N_PAD rows for guard-free GEMM)
  char* w = (char*)d_ws;
  auto carve = [&](size_t bytes) -> char* {
    char* p = w; w += (bytes + 255) & ~(size_t)255; return p;
  };
  f16*   xh     = (f16*)  carve((size_t)N_PAD * HID * sizeof(f16));
  float* xl     = (float*)carve((size_t)N_PAD * HID * sizeof(float));
  float* xr     = (float*)carve((size_t)N_PAD * HID * sizeof(float));
  float* acc0   = (float*)carve((size_t)N_NODES * HID * sizeof(float));
  float* logits = (float*)carve((size_t)EF * 4 * sizeof(float));
  float* lmax   = (float*)carve((size_t)N_NODES * 4 * sizeof(float));
  float* denom  = (float*)carve((size_t)N_NODES * 4 * sizeof(float));
  float* deg    = (float*)carve((size_t)N_NODES * sizeof(float));
  float* easum  = (float*)carve((size_t)N_NODES * sizeof(float));
  f16*   WlH    = (f16*)  carve((size_t)HID * HID * sizeof(f16));
  f16*   WrH    = (f16*)  carve((size_t)HID * HID * sizeof(f16));

  const int GRID_NC  = (N_NODES * HID) / 256;        // 25000 (exact)
  const int GRID_E   = (E_EDGES + 255) / 256;        // 3125
  const int GRID_EFW = (EF + 7) / 8;                 // one wave per edge
  const int GRID_EF4 = (int)(((long)EF * 4 + 255) / 256);
  const int GRID_W   = (2 * HID * HID + 255) / 256;  // 128
  const dim3 GEMM_GRID(N_PAD / 64, 2);               // 782 x {l,r}

  for (int layer = 0; layer < 2; ++layer) {
    float* acc = (layer == 0) ? acc0 : (float*)d_out;

    gat_init<<<GRID_NC, 256, 0, stream>>>(x, xh, bb[layer], acc, lmax, denom,
                                          deg, easum, layer == 0 ? 1 : 0);
    gat_degree<<<GRID_E, 256, 0, stream>>>(dst, ew, deg, easum);
    gat_convw<<<GRID_W, 256, 0, stream>>>(Wl[layer], Wr[layer], WlH, WrH);
    gat_gemm_wmma<<<GEMM_GRID, 256, 0, stream>>>(xh, WlH, WrH, bl[layer],
                                                 br[layer], xl, xr);
    gat_logits_max<<<GRID_EFW, 256, 0, stream>>>(src, dst, ew, deg, easum,
                                                 xl, xr, We[layer], att[layer],
                                                 logits, lmax);
    gat_expsum<<<GRID_EF4, 256, 0, stream>>>(dst, lmax, logits, denom);
    gat_aggregate<<<GRID_EFW, 256, 0, stream>>>(src, dst, logits, denom, xl, acc);

    if (layer == 0)
      gat_prelu_conv<<<GRID_NC, 256, 0, stream>>>(acc, prelu_a, xh);
  }
}
